// FineGrainLoss_54168127537809
// MI455X (gfx1250) — compile-verified, matrix operand order checked by reading, not run on hardware
//
#include <hip/hip_runtime.h>

typedef __attribute__((ext_vector_type(16))) __bf16 v16bf;
typedef __attribute__((ext_vector_type(8)))  float  v8f;

#define B_SZ    96
#define N_Q     196
#define N_R     77
#define D_DIM   512
#define M_PAD   208                 // image tokens padded to 13*16
#define N_PAD   80                  // text tokens padded to 5*16
#define K_CH    128                 // K-chunk staged in LDS
#define LDS_STR 136                 // 128 + 8 ushort pad -> conflict-free b128 LDS reads
#define M_TILES (M_PAD / 16)        // 13
#define N_TILES (N_PAD / 16)        // 5
#define TILES   (M_TILES * N_TILES) // 65
#define WAVES   8
#define TPW     9                   // ceil(65/8) tiles per wave

#if __has_builtin(__builtin_amdgcn_global_load_async_to_lds_b128)
#define HAVE_ASYNC_LDS 1
// Builtin wants pointers to GCC-style <4 x i32> vectors in AS1 (global) / AS3 (LDS)
// (exact spelling from the clang diagnostic: __vector_size__(4*sizeof(int)) int).
typedef int v4i_async __attribute__((vector_size(16)));
typedef __attribute__((address_space(1))) v4i_async gas1_v4i;
typedef __attribute__((address_space(3))) v4i_async gas3_v4i;
#endif

// Copy 16 bytes global -> LDS. Async path uses the CDNA5 ASYNCcnt-tracked DMA
// (no VGPR round-trip); fallback is a plain load+store pair.
__device__ __forceinline__ void copy16_g2l(const unsigned short* g, unsigned short* l) {
#if defined(HAVE_ASYNC_LDS)
  // Low 32 bits of a generic LDS pointer are the LDS byte offset (ISA 10.2
  // aperture rule), so the uintptr_t->AS3 truncation is exact. AS1 is identity.
  __builtin_amdgcn_global_load_async_to_lds_b128(
      (gas1_v4i*)(uintptr_t)g, (gas3_v4i*)(uintptr_t)l, 0, 0);
#else
  *(uint4*)l = *(const uint4*)g;
#endif
}

__device__ __forceinline__ void wait_async_fills() {
#if defined(HAVE_ASYNC_LDS)
#if __has_builtin(__builtin_amdgcn_s_wait_asynccnt)
  __builtin_amdgcn_s_wait_asynccnt(0);
#else
  asm volatile("s_wait_asynccnt 0" ::: "memory");
#endif
#endif
}

// ---- monotone float <-> uint key so LDS u32 atomicMax implements float max ----
__device__ __forceinline__ unsigned f2key(float x) {
  unsigned u = __float_as_uint(x);
  return (u & 0x80000000u) ? ~u : (u | 0x80000000u);
}
__device__ __forceinline__ float key2f(unsigned k) {
  return __uint_as_float((k & 0x80000000u) ? (k & 0x7fffffffu) : ~k);
}

// ---- fp32 -> (hi, lo) bf16 split with RNE, padded token rows set to zero ----
__global__ void split_pad_kernel(const float* __restrict__ src,
                                 unsigned short* __restrict__ hi,
                                 unsigned short* __restrict__ lo,
                                 int rows, int rowsPad, int total) {
  int i = blockIdx.x * blockDim.x + threadIdx.x;
  if (i >= total) return;
  int k   = i & (D_DIM - 1);
  int t   = i >> 9;                // / D_DIM
  int row = t % rowsPad;
  int b   = t / rowsPad;
  float x = 0.f;
  if (row < rows) x = src[((size_t)b * rows + row) * D_DIM + k];
  unsigned u = __float_as_uint(x);
  unsigned short h = (unsigned short)((u + 0x7fffu + ((u >> 16) & 1u)) >> 16);
  float hf = __uint_as_float((unsigned)h << 16);
  float l  = x - hf;
  unsigned ul = __float_as_uint(l);
  unsigned short lw = (unsigned short)((ul + 0x7fffu + ((ul >> 16) & 1u)) >> 16);
  hi[i] = h;
  lo[i] = lw;
}

// ---- one workgroup per (image-batch bi, text-batch bt) pair ----
__global__ __launch_bounds__(256) void pair_scores_kernel(
    const unsigned short* __restrict__ imgHi, const unsigned short* __restrict__ imgLo,
    const unsigned short* __restrict__ txtHi, const unsigned short* __restrict__ txtLo,
    float* __restrict__ i2t, float* __restrict__ t2i) {
  __shared__ unsigned short sAh[M_PAD * LDS_STR];
  __shared__ unsigned short sAl[M_PAD * LDS_STR];
  __shared__ unsigned short sBh[N_PAD * LDS_STR];
  __shared__ unsigned short sBl[N_PAD * LDS_STR];
  __shared__ unsigned rowMaxK[M_PAD];
  __shared__ unsigned colMaxK[N_PAD];
  __shared__ float red[64];

  const int bi = blockIdx.x / B_SZ;
  const int bt = blockIdx.x % B_SZ;
  const int tid    = threadIdx.x;
  const int wave   = tid >> 5;
  const int lane   = tid & 31;
  const int laneLo = lane & 15;
  const int laneHi = lane >> 4;

  for (int i = tid; i < M_PAD; i += 256) rowMaxK[i] = 0u;  // key for "-inf"
  for (int i = tid; i < N_PAD; i += 256) colMaxK[i] = 0u;

  const unsigned short* __restrict__ Ah = imgHi + (size_t)bi * (M_PAD * D_DIM);
  const unsigned short* __restrict__ Al = imgLo + (size_t)bi * (M_PAD * D_DIM);
  const unsigned short* __restrict__ Bh = txtHi + (size_t)bt * (N_PAD * D_DIM);
  const unsigned short* __restrict__ Bl = txtLo + (size_t)bt * (N_PAD * D_DIM);

  v8f acc[TPW];
#pragma unroll
  for (int j = 0; j < TPW; ++j) { v8f z = {}; acc[j] = z; }

  for (int k0 = 0; k0 < D_DIM; k0 += K_CH) {
    __syncthreads();  // previous chunk consumed (and rowMax init visible)
    // stage A/B (hi+lo) K-chunk into LDS, 16B per thread-step (async DMA if available)
    for (int idx = tid; idx < M_PAD * (K_CH / 8); idx += 256) {
      int row = idx >> 4;          // K_CH/8 == 16 groups per row
      int c8  = (idx & 15) << 3;
      size_t g = (size_t)row * D_DIM + k0 + c8;
      copy16_g2l(&Ah[g], &sAh[row * LDS_STR + c8]);
      copy16_g2l(&Al[g], &sAl[row * LDS_STR + c8]);
    }
    for (int idx = tid; idx < N_PAD * (K_CH / 8); idx += 256) {
      int row = idx >> 4;
      int c8  = (idx & 15) << 3;
      size_t g = (size_t)row * D_DIM + k0 + c8;
      copy16_g2l(&Bh[g], &sBh[row * LDS_STR + c8]);
      copy16_g2l(&Bl[g], &sBl[row * LDS_STR + c8]);
    }
    wait_async_fills();
    __syncthreads();

#pragma unroll
    for (int j = 0; j < TPW; ++j) {
      const int tile = wave + WAVES * j;
      if (tile < TILES) {
        const int mt = tile / N_TILES;
        const int nt = tile % N_TILES;
        // A frag: lane = row M (mod 16); lanes 16-31 hold K+8 halves (ISA 16-bit A layout)
        const unsigned short* sa_h = &sAh[(mt * 16 + laneLo) * LDS_STR + laneHi * 8];
        const unsigned short* sa_l = &sAl[(mt * 16 + laneLo) * LDS_STR + laneHi * 8];
        const unsigned short* sb_h = &sBh[(nt * 16 + laneLo) * LDS_STR + laneHi * 8];
        const unsigned short* sb_l = &sBl[(nt * 16 + laneLo) * LDS_STR + laneHi * 8];
#pragma unroll
        for (int kc = 0; kc < K_CH; kc += 32) {
          union { v16bf v; uint4 u[2]; } fah, fal, fbh, fbl;
          fah.u[0] = *(const uint4*)(sa_h + kc);
          fah.u[1] = *(const uint4*)(sa_h + kc + 16);
          fal.u[0] = *(const uint4*)(sa_l + kc);
          fal.u[1] = *(const uint4*)(sa_l + kc + 16);
          fbh.u[0] = *(const uint4*)(sb_h + kc);
          fbh.u[1] = *(const uint4*)(sb_h + kc + 16);
          fbl.u[0] = *(const uint4*)(sb_l + kc);
          fbl.u[1] = *(const uint4*)(sb_l + kc + 16);
          // hi*hi + hi*lo + lo*hi  (near-fp32 dot from bf16 WMMA)
          acc[j] = __builtin_amdgcn_wmma_f32_16x16x32_bf16(false, fah.v, false, fbh.v,
                                                           (short)0, acc[j], false, false);
          acc[j] = __builtin_amdgcn_wmma_f32_16x16x32_bf16(false, fah.v, false, fbl.v,
                                                           (short)0, acc[j], false, false);
          acc[j] = __builtin_amdgcn_wmma_f32_16x16x32_bf16(false, fal.v, false, fbh.v,
                                                           (short)0, acc[j], false, false);
        }
      }
    }
  }

  // per-element max scatter into row/col maxima (mask out padded q/r)
#pragma unroll
  for (int j = 0; j < TPW; ++j) {
    const int tile = wave + WAVES * j;
    if (tile < TILES) {
      const int mt = tile / N_TILES;
      const int nt = tile % N_TILES;
      const int r = nt * 16 + laneLo;        // C/D layout: lane%16 = column N
      if (r < N_R) {
#pragma unroll
        for (int i = 0; i < 8; ++i) {
          const int q = mt * 16 + i + 8 * laneHi;  // VGPR i, lanes 16-31 -> M+8
          if (q < N_Q) {
            const unsigned key = f2key(acc[j][i]);
            atomicMax(&rowMaxK[q], key);
            atomicMax(&colMaxK[r], key);
          }
        }
      }
    }
  }
  __syncthreads();

  // deterministic tree reductions: mean of row maxima, mean of col maxima
  float part = 0.f;
  if (tid < 64) {
    for (int q = tid; q < N_Q; q += 64) part += key2f(rowMaxK[q]);
    red[tid] = part;
  }
  __syncthreads();
  for (int s = 32; s >= 1; s >>= 1) {
    if (tid < s) red[tid] += red[tid + s];
    __syncthreads();
  }
  if (tid == 0) i2t[bi * B_SZ + bt] = red[0] * (1.0f / N_Q);
  __syncthreads();
  part = 0.f;
  if (tid < 64) {
    for (int r = tid; r < N_R; r += 64) part += key2f(colMaxK[r]);
    red[tid] = part;
  }
  __syncthreads();
  for (int s = 32; s >= 1; s >>= 1) {
    if (tid < s) red[tid] += red[tid + s];
    __syncthreads();
  }
  if (tid == 0) t2i[bt * B_SZ + bi] = red[0] * (1.0f / N_R);
}

// ---- cross-entropy with arange labels on both 96x96 logit matrices ----
__global__ __launch_bounds__(128) void ce_loss_kernel(const float* __restrict__ i2t,
                                                      const float* __restrict__ t2i,
                                                      float* __restrict__ out) {
  __shared__ float red[128];
  const int t = threadIdx.x;
  float acc = 0.f;
  if (t < B_SZ) {
    for (int which = 0; which < 2; ++which) {
      const float* row = (which == 0 ? i2t : t2i) + t * B_SZ;
      float m = -3.4e38f;
      for (int j = 0; j < B_SZ; ++j) m = fmaxf(m, row[j]);
      float sum = 0.f;
      for (int j = 0; j < B_SZ; ++j) sum += __expf(row[j] - m);
      const float lse = m + __logf(sum);
      acc += lse - row[t];  // = -log_softmax(row)[t]
    }
  }
  red[t] = acc;
  __syncthreads();
  for (int s = 64; s >= 1; s >>= 1) {
    if (t < s) red[t] += red[t + s];
    __syncthreads();
  }
  if (t == 0) out[0] = 0.5f * red[0] / (float)B_SZ;
}

extern "C" void kernel_launch(void* const* d_in, const int* in_sizes, int n_in,
                              void* d_out, int out_size, void* d_ws, size_t ws_size,
                              hipStream_t stream) {
  const float* img = (const float*)d_in[0];  // [96, 196, 512] fp32
  const float* txt = (const float*)d_in[1];  // [96, 77, 512] fp32

  const size_t imgPad = (size_t)B_SZ * M_PAD * D_DIM;  // 10,223,616
  const size_t txtPad = (size_t)B_SZ * N_PAD * D_DIM;  //  3,932,160
  unsigned short* imgHi = (unsigned short*)d_ws;
  unsigned short* imgLo = imgHi + imgPad;
  unsigned short* txtHi = imgLo + imgPad;
  unsigned short* txtLo = txtHi + txtPad;
  float* i2t = (float*)(txtLo + txtPad);     // byte offset multiple of 16
  float* t2i = i2t + B_SZ * B_SZ;

  {
    const int total = (int)imgPad;
    split_pad_kernel<<<(total + 255) / 256, 256, 0, stream>>>(img, imgHi, imgLo,
                                                              N_Q, M_PAD, total);
  }
  {
    const int total = (int)txtPad;
    split_pad_kernel<<<(total + 255) / 256, 256, 0, stream>>>(txt, txtHi, txtLo,
                                                              N_R, N_PAD, total);
  }
  pair_scores_kernel<<<B_SZ * B_SZ, 256, 0, stream>>>(imgHi, imgLo, txtHi, txtLo,
                                                      i2t, t2i);
  ce_loss_kernel<<<1, 128, 0, stream>>>(i2t, t2i, (float*)d_out);
}
